// MultiHeadAttention_62835371540559
// MI455X (gfx1250) — matrix operand order repaired and checked by reading, not compiled, not run on hardware
//
#include <hip/hip_runtime.h>

// ---------------------------------------------------------------------------
// MI455X (gfx1250) multi-head attention, bf16 WMMA (16x16x32), f32 accum.
// QKV proj GEMMs -> flash-attention (softmax over the query axis, faithful to
// the reference) -> output proj GEMM. Matrix math via
// v_wmma_f32_16x16x32_bf16; flash-attention tiles staged into LDS with
// global_load_async_to_lds_b128 (ASYNCcnt-tracked, no VGPR round trip).
// ---------------------------------------------------------------------------

typedef __bf16 bf16;
typedef __attribute__((ext_vector_type(8)))  __bf16 bf16x8;
typedef __attribute__((ext_vector_type(16))) __bf16 bf16x16;
typedef __attribute__((ext_vector_type(8)))  float  f32x8;

#define B_  2
#define S_  2048
#define H_  16
#define HD_ 64
#define DM_ 1024

// A-fragment (16x32 bf16, MxK): lane l<16 -> row l, K {0..7,16..23};
// lane l>=16 -> row l-16, K {8..15,24..31}. Row-major LDS rows: each lane
// reads two 16B chunks at +0 and +32B from (row*stride + (l>=16?8:0)).
__device__ __forceinline__ bf16x16 frag_a(const bf16* p) {
  union { bf16x16 v; bf16x8 h[2]; } u;
  u.h[0] = *(const bf16x8*)(p);
  u.h[1] = *(const bf16x8*)(p + 16);
  return u.v;
}
// B-fragment (32x16 bf16, KxN) from a [n][k]-transposed LDS tile: lane l
// covers column n=l%16, K (l<16 ? {0..7,16..23} : {8..15,24..31}).
__device__ __forceinline__ bf16x16 frag_b(const bf16* p) {
  union { bf16x16 v; bf16x8 h[2]; } u;
  u.h[0] = *(const bf16x8*)(p);
  u.h[1] = *(const bf16x8*)(p + 16);
  return u.v;
}
__device__ __forceinline__ f32x8 wmma_bf16(bf16x16 a, bf16x16 b, f32x8 c) {
  return __builtin_amdgcn_wmma_f32_16x16x32_bf16(false, a, false, b,
                                                 (short)0, c, false, false);
}

// Async 16B global->LDS copy (GV mode, per-lane). The LDS byte offset is the
// low 32 bits of the flat shared-aperture address. Tracked on ASYNCcnt.
__device__ __forceinline__ void async_ld_b128(bf16* lds_dst, const bf16* gsrc) {
  const unsigned lds_off = (unsigned)(uintptr_t)lds_dst;
  asm volatile("global_load_async_to_lds_b128 %0, %1, off"
               :: "v"(lds_off), "v"(gsrc)
               : "memory");
}
__device__ __forceinline__ void wait_asynccnt0() {
  asm volatile("s_wait_asynccnt 0x0" ::: "memory");
}

// ---------------------------------------------------------------------------
// GEMM: C[4096,1024] = A[4096,1024] @ W[1024,1024] + bias
// OMODE 0: out bf16 [B,H,S,HD]   (Qp / Kp)
// OMODE 1: out bf16 [B,H,HD,S]   (Vp transposed, so attn B-frags are rows)
// OMODE 2: out f32  [M,N]        (final output)
// ---------------------------------------------------------------------------
template <bool ABF16, int OMODE>
__global__ __launch_bounds__(256) void gemm_wmma(
    const void* __restrict__ Ain, const float* __restrict__ W,
    const float* __restrict__ bias, void* __restrict__ Out) {
  constexpr int K = 1024, N = 1024;
  __shared__ __align__(32) bf16 sA[128 * 40];  // [m][k], stride 40 (pad)
  __shared__ __align__(32) bf16 sB[64 * 40];   // [n][k], stride 40 (pad)

  const int tid  = threadIdx.x;
  const int n0   = blockIdx.x * 64;
  const int m0   = blockIdx.y * 128;
  const int lane = tid & 31, wave = tid >> 5;
  const int wm = wave >> 1, wn = wave & 1;       // 4x2 wave grid
  const int l16 = lane & 15, lh = lane >> 4;
  const int ar = tid >> 1, ah = tid & 1;         // A staging: 2 threads/row
  const int bk = tid & 31, bn = (tid >> 5) * 8;  // B staging

  const float* Af = (const float*)Ain;
  const bf16*  Ab = (const bf16*)Ain;

  f32x8 acc[2][2] = {};

  for (int k0 = 0; k0 < K; k0 += 32) {
    __syncthreads();
    // --- A tile 128x32 -> LDS (convert f32->bf16 on the fly if needed)
    if constexpr (!ABF16) {
      const float* ap = Af + (size_t)(m0 + ar) * K + k0 + ah * 16;
      bf16x8 o0, o1;
#pragma unroll
      for (int j = 0; j < 8; ++j) { o0[j] = (bf16)ap[j]; o1[j] = (bf16)ap[j + 8]; }
      *(bf16x8*)&sA[ar * 40 + ah * 16]     = o0;
      *(bf16x8*)&sA[ar * 40 + ah * 16 + 8] = o1;
      if (k0 + 32 < K) __builtin_prefetch(ap + 32, 0, 1);
    } else {
      const bf16* ap = Ab + (size_t)(m0 + ar) * K + k0 + ah * 16;
      *(bf16x8*)&sA[ar * 40 + ah * 16]     = *(const bf16x8*)ap;
      *(bf16x8*)&sA[ar * 40 + ah * 16 + 8] = *(const bf16x8*)(ap + 8);
      if (k0 + 32 < K) __builtin_prefetch(ap + 32, 0, 1);
    }
    // --- B tile 32x64 -> LDS transposed [n][k]
    {
      const float* wp = W + (size_t)(k0 + bk) * N + n0 + bn;
#pragma unroll
      for (int j = 0; j < 8; ++j) sB[(bn + j) * 40 + bk] = (bf16)wp[j];
    }
    __syncthreads();

    const bf16x16 a0 = frag_a(&sA[(wm * 32 + 0  + l16) * 40 + lh * 8]);
    const bf16x16 a1 = frag_a(&sA[(wm * 32 + 16 + l16) * 40 + lh * 8]);
    const bf16x16 b0 = frag_b(&sB[(wn * 32 + 0  + l16) * 40 + lh * 8]);
    const bf16x16 b1 = frag_b(&sB[(wn * 32 + 16 + l16) * 40 + lh * 8]);
    acc[0][0] = wmma_bf16(a0, b0, acc[0][0]);
    acc[0][1] = wmma_bf16(a0, b1, acc[0][1]);
    acc[1][0] = wmma_bf16(a1, b0, acc[1][0]);
    acc[1][1] = wmma_bf16(a1, b1, acc[1][1]);
  }

  // --- epilogue: bias + layout-specific store.
  // C/D layout: lanes 0-15 hold rows 0-7 (vgpr r -> M=r), lanes 16-31 rows 8-15.
#pragma unroll
  for (int i = 0; i < 2; ++i)
#pragma unroll
    for (int j = 0; j < 2; ++j) {
      const int n = n0 + wn * 32 + j * 16 + l16;
      const float bv = bias[n];
#pragma unroll
      for (int r = 0; r < 8; ++r) {
        const int m = m0 + wm * 32 + i * 16 + r + lh * 8;
        const float val = acc[i][j][r] + bv;
        if constexpr (OMODE == 2) {
          ((float*)Out)[(size_t)m * N + n] = val;
        } else {
          const int b = m >> 11, s = m & (S_ - 1);
          const int h = n >> 6,  d = n & (HD_ - 1);
          size_t addr;
          if constexpr (OMODE == 0)
            addr = (((size_t)(b * H_ + h)) * S_ + s) * HD_ + d;
          else
            addr = (((size_t)(b * H_ + h)) * HD_ + d) * S_ + s;
          ((bf16*)Out)[addr] = (bf16)val;
        }
      }
    }
}

// ---------------------------------------------------------------------------
// Flash attention per (b,h): scores[c,q] = Kp[c,:]·Qp[q,:], softmax over q,
// O[c,:] = sum_q attn[c,q] * Vp[q,:]. One block handles 128 c-rows; each of
// the 8 waves owns 16 rows; q streamed in blocks of 64 with online softmax.
// All global->LDS tile traffic uses async-to-LDS (ASYNCcnt).
// ---------------------------------------------------------------------------
__global__ __launch_bounds__(256) void flash_attn(
    const bf16* __restrict__ Kp, const bf16* __restrict__ Qp,
    const bf16* __restrict__ VpT, bf16* __restrict__ O) {
  __shared__ __align__(32) bf16 sK[128 * 72];     // [c][d]
  __shared__ __align__(32) bf16 sQ[64 * 72];      // [q][d]  (== B^T for scores)
  __shared__ __align__(32) bf16 sV[64 * 72];      // [d][q]  (== B^T for P·V)
  __shared__ __align__(32) bf16 sP[8 * 16 * 72];  // per-wave P scratch [c][q]

  const int tid  = threadIdx.x;
  const int lane = tid & 31, wave = tid >> 5;
  const int l16 = lane & 15, lh = lane >> 4;
  const int bh = blockIdx.y;            // b*H + h
  const int c0 = blockIdx.x * 128;
  const int b = bh >> 4, h = bh & 15;

  // K tile (128 x 64), loaded once, asynchronously.
  {
    const int r = tid >> 1, half = tid & 1;
    const bf16* kp = Kp + ((size_t)bh * S_ + c0 + r) * HD_ + half * 32;
#pragma unroll
    for (int cc = 0; cc < 4; ++cc)
      async_ld_b128(&sK[r * 72 + half * 32 + cc * 8], kp + cc * 8);
  }

  f32x8 oacc[4] = {};
  float m_run[8], l_run[8];
#pragma unroll
  for (int r = 0; r < 8; ++r) { m_run[r] = -1e30f; l_run[r] = 0.0f; }

  bf16* spw = &sP[wave * 16 * 72];
  const int qr = tid & 63, qch = tid >> 6;   // Q/V staging: 64 rows x 4 chunks

  for (int q0 = 0; q0 < S_; q0 += 64) {
    __syncthreads();  // all waves done reading the previous sQ/sV
    {
      const bf16* qp = Qp + ((size_t)bh * S_ + q0 + qr) * HD_ + qch * 16;
      async_ld_b128(&sQ[qr * 72 + qch * 16],     qp);
      async_ld_b128(&sQ[qr * 72 + qch * 16 + 8], qp + 8);
      const bf16* vp = VpT + ((size_t)bh * HD_ + qr) * S_ + q0 + qch * 16;
      async_ld_b128(&sV[qr * 72 + qch * 16],     vp);
      async_ld_b128(&sV[qr * 72 + qch * 16 + 8], vp + 8);
    }
    wait_asynccnt0();   // our async transfers (incl. sK on iter 0) landed
    __syncthreads();    // ... and everyone else's

    // scores: (16 c-rows) x (64 q-cols), K-dim = HD (two k-steps of 32)
    f32x8 sacc[4] = {};
#pragma unroll
    for (int ks = 0; ks < 2; ++ks) {
      const bf16x16 a = frag_a(&sK[(wave * 16 + l16) * 72 + ks * 32 + lh * 8]);
#pragma unroll
      for (int j = 0; j < 4; ++j) {
        const bf16x16 bq = frag_b(&sQ[(j * 16 + l16) * 72 + ks * 32 + lh * 8]);
        sacc[j] = wmma_bf16(a, bq, sacc[j]);
      }
    }

    // Online softmax across q (the N axis). Each vgpr r is row r (+8 for
    // lanes 16-31); XOR masks 1,2,4,8 reduce across the 16 N-lanes of a row.
#pragma unroll
    for (int r = 0; r < 8; ++r) {
      float mv = fmaxf(fmaxf(sacc[0][r], sacc[1][r]),
                       fmaxf(sacc[2][r], sacc[3][r]));
#pragma unroll
      for (int msk = 1; msk < 16; msk <<= 1)
        mv = fmaxf(mv, __shfl_xor(mv, msk, 32));
      const float mn = fmaxf(m_run[r], mv);
      const float sc = __expf(m_run[r] - mn);
      m_run[r] = mn;
      float rs = 0.0f;
#pragma unroll
      for (int j = 0; j < 4; ++j) {
        const float p = __expf(sacc[j][r] - mn);
        sacc[j][r] = p;
        rs += p;
      }
#pragma unroll
      for (int msk = 1; msk < 16; msk <<= 1) rs += __shfl_xor(rs, msk, 32);
      l_run[r] = l_run[r] * sc + rs;
#pragma unroll
      for (int j = 0; j < 4; ++j) oacc[j][r] *= sc;
      // stage P (bf16) into per-wave LDS, row-major [c][q] -> A-frag layout
#pragma unroll
      for (int j = 0; j < 4; ++j)
        spw[(r + lh * 8) * 72 + j * 16 + l16] = (bf16)sacc[j][r];
    }

    // O += P(16x64) @ V(64x64). LDS ops are in-order within a wave, so the
    // P stores above are visible to these fragment loads without a barrier.
#pragma unroll
    for (int ks = 0; ks < 2; ++ks) {
      const bf16x16 a = frag_a(&spw[l16 * 72 + ks * 32 + lh * 8]);
#pragma unroll
      for (int j = 0; j < 4; ++j) {
        const bf16x16 bv = frag_b(&sV[(j * 16 + l16) * 72 + ks * 32 + lh * 8]);
        oacc[j] = wmma_bf16(a, bv, oacc[j]);
      }
    }
  }

  // Normalize and store O as bf16 [B, S, H*HD] for the output projection.
#pragma unroll
  for (int r = 0; r < 8; ++r) {
    const float inv = 1.0f / l_run[r];
    const int c = c0 + wave * 16 + r + lh * 8;
#pragma unroll
    for (int j = 0; j < 4; ++j) {
      const int n = h * HD_ + j * 16 + l16;
      O[((size_t)(b * S_ + c)) * DM_ + n] = (bf16)(oacc[j][r] * inv);
    }
  }
}

// ---------------------------------------------------------------------------
extern "C" void kernel_launch(void* const* d_in, const int* in_sizes, int n_in,
                              void* d_out, int out_size, void* d_ws,
                              size_t ws_size, hipStream_t stream) {
  (void)in_sizes; (void)n_in; (void)out_size; (void)ws_size;
  const float* k  = (const float*)d_in[0];
  const float* q  = (const float*)d_in[1];
  const float* v  = (const float*)d_in[2];
  const float* Wk = (const float*)d_in[3];
  const float* bk = (const float*)d_in[4];
  const float* Wq = (const float*)d_in[5];
  const float* bq = (const float*)d_in[6];
  const float* Wv = (const float*)d_in[7];
  const float* bv = (const float*)d_in[8];
  const float* Wo = (const float*)d_in[9];
  const float* bo = (const float*)d_in[10];
  float* out = (float*)d_out;

  bf16* ws = (bf16*)d_ws;
  const size_t PROJ = (size_t)B_ * H_ * S_ * HD_;  // 4,194,304 elements
  bf16* Qp  = ws;               // [B,H,S,HD] bf16
  bf16* Kp  = Qp  + PROJ;       // [B,H,S,HD] bf16
  bf16* VpT = Kp  + PROJ;       // [B,H,HD,S] bf16
  bf16* Oat = VpT + PROJ;       // [B,S,DM]   bf16

  const dim3 blk(256);
  const dim3 g_gemm(DM_ / 64, (B_ * S_) / 128);   // (16, 32)
  const dim3 g_attn(S_ / 128, B_ * H_);           // (16, 32)

  gemm_wmma<false, 0><<<g_gemm, blk, 0, stream>>>(q, Wq, bq, Qp);
  gemm_wmma<false, 0><<<g_gemm, blk, 0, stream>>>(k, Wk, bk, Kp);
  gemm_wmma<false, 1><<<g_gemm, blk, 0, stream>>>(v, Wv, bv, VpT);
  flash_attn<<<g_attn, blk, 0, stream>>>(Kp, Qp, VpT, Oat);
  gemm_wmma<true, 2><<<g_gemm, blk, 0, stream>>>(Oat, Wo, bo, out);
}
